// FalconAttention_46308337385925
// MI455X (gfx1250) — matrix operand order repaired and checked by reading, not compiled, not run on hardware
//
#include <hip/hip_runtime.h>
#include <hip/hip_bf16.h>
#include <stdint.h>

// ---------------------------------------------------------------------------
// Falcon attention block for MI455X (gfx1250, wave32, WMMA + TDM staging).
// ---------------------------------------------------------------------------

typedef __bf16 bf16;
typedef __attribute__((ext_vector_type(16))) __bf16 v16bf;
typedef __attribute__((ext_vector_type(8)))  __bf16 v8bf;
typedef __attribute__((ext_vector_type(8)))  float  v8f;
typedef __attribute__((ext_vector_type(4)))  unsigned int u32x4;
typedef __attribute__((ext_vector_type(4)))  int i32x4;
typedef __attribute__((ext_vector_type(8)))  int i32x8;

#define WMMA_BF16(a, b, c) \
  __builtin_amdgcn_wmma_f32_16x16x32_bf16(false, (a), false, (b), (short)0, (c), false, false)

// Tensor Data Mover availability (device pass only; host pass lacks builtin).
#if defined(__has_builtin)
#if __has_builtin(__builtin_amdgcn_tensor_load_to_lds) && \
    __has_builtin(__builtin_amdgcn_s_wait_tensorcnt)
#define HAVE_TDM 1
#endif
#endif
#ifndef HAVE_TDM
#define HAVE_TDM 0
#endif

// TDM writes LDS behind the compiler's back.  Passing the LDS object's
// address INTO a memory-clobbering asm makes it escape, so alias analysis
// must assume the asm may write it -> the subsequent ds_loads are kept.
__device__ __forceinline__ void lds_publish(const void* p) {
  asm volatile("" : : "v"(p) : "memory");
}

constexpr int B_    = 2;
constexpr int S_    = 2048;
constexpr int H_    = 4096;
constexpr int NH    = 32;
constexpr int NKVH  = 8;
constexpr int HD    = 128;
constexpr int QKV_N = (NKVH * 2 + NH) * HD;   // 6144
constexpr int M_TOK = B_ * S_;                // 4096

// ---------------------------------------------------------------------------
// Fragment helpers (CDNA5 16x16x32 bf16 striping, wave32).
// A 16x32: lanes 0-15: row=lane, K in {0..7, 16..23}; lanes 16-31: K+8.
// B 32x16: lanes 0-15: col=lane, K in {0..7, 16..23}; lanes 16-31: col=lane-16, K+8.
// ---------------------------------------------------------------------------
__device__ __forceinline__ v16bf make_v16(v8bf lo, v8bf hi) {
  v16bf r;
#pragma unroll
  for (int i = 0; i < 8; ++i) { r[i] = lo[i]; r[i + 8] = hi[i]; }
  return r;
}

__device__ __forceinline__ v16bf load_a_frag_g(const bf16* __restrict__ src, int ld, int k0) {
  const int lane = threadIdx.x & 31;
  const int row  = lane & 15;
  const int koff = (lane >> 4) << 3;
  const bf16* p  = src + (size_t)row * ld + k0 + koff;
  v8bf lo = *reinterpret_cast<const v8bf*>(p);
  v8bf hi = *reinterpret_cast<const v8bf*>(p + 16);
  return make_v16(lo, hi);
}

__device__ __forceinline__ v16bf load_b_frag_g(const bf16* __restrict__ src, int ld, int k0) {
  const int lane = threadIdx.x & 31;
  const int col  = lane & 15;
  const int koff = (lane >> 4) << 3;
  const bf16* p  = src + (size_t)col * ld + k0 + koff;
  v8bf lo = *reinterpret_cast<const v8bf*>(p);
  v8bf hi = *reinterpret_cast<const v8bf*>(p + 16);
  return make_v16(lo, hi);
}

// ---------------------------------------------------------------------------
// TDM 2-D tile load descriptor (CDNA5 ISA).
// group0: count=1, lds_addr, global_addr[56:0], type=2.
// group1: data_size=2B, tensor_dim0/1, tile_dim0/1, tensor_dim0_stride.
// ---------------------------------------------------------------------------
#if HAVE_TDM
__device__ __forceinline__ void tdm_load_2d(unsigned lds_off, const void* gaddr,
                                            unsigned tensor_d0, unsigned tensor_d1,
                                            unsigned tile_d0, unsigned tile_d1,
                                            unsigned stride_d0) {
  const unsigned long long ga = (unsigned long long)gaddr;
  u32x4 g0;
  g0[0] = 1u;                                             // count=1, flags 0
  g0[1] = lds_off;                                        // lds_addr (bytes)
  g0[2] = (unsigned)ga;                                   // global_addr[31:0]
  g0[3] = (unsigned)((ga >> 32) & 0x01FFFFFFu) | (2u << 30);  // [56:32] | type=2
  i32x8 g1;
  g1[0] = (int)(1u << 16);                                // data_size = 1 -> 2 bytes
  g1[1] = (int)((tensor_d0 & 0xFFFFu) << 16);             // tensor_dim0[15:0]
  g1[2] = (int)(((tensor_d0 >> 16) & 0xFFFFu) | ((tensor_d1 & 0xFFFFu) << 16));
  g1[3] = (int)(((tensor_d1 >> 16) & 0xFFFFu) | (tile_d0 << 16));  // tile_dim0
  g1[4] = (int)(tile_d1 & 0xFFFFu);                       // tile_dim1, tile_dim2=0
  g1[5] = (int)stride_d0;                                 // dim0 stride [31:0]
  g1[6] = 0;                                              // stride[47:32], dim1 stride
  g1[7] = 0;
  i32x4 z4 = {0, 0, 0, 0};
#if __clang_major__ >= 23
  i32x8 z8 = {0, 0, 0, 0, 0, 0, 0, 0};
  __builtin_amdgcn_tensor_load_to_lds(g0, g1, z4, z4, z8, 0);
#else
  __builtin_amdgcn_tensor_load_to_lds(g0, g1, z4, z4, 0);
#endif
}
#endif

// ---------------------------------------------------------------------------
// fp32 -> bf16 conversion (n multiple of 4)
// ---------------------------------------------------------------------------
__global__ __launch_bounds__(256) void cvt_f32_bf16(const float* __restrict__ in,
                                                    bf16* __restrict__ out, int n) {
  int i = (blockIdx.x * blockDim.x + threadIdx.x) * 4;
  if (i + 3 < n) {
    float4 v = *reinterpret_cast<const float4*>(in + i);
    out[i]     = (bf16)v.x;
    out[i + 1] = (bf16)v.y;
    out[i + 2] = (bf16)v.z;
    out[i + 3] = (bf16)v.w;
  }
}

// ---------------------------------------------------------------------------
// NT GEMM: C[M,N] = A[M,K] * W[N,K]^T, bf16 in, f32 out.
// 256 threads = 8 waves (2x4); block tile 64x256; wave tile 32x64;
// K in 32-deep LDS slices, double-buffered.  Staging: TDM if available,
// else register-staged global loads (all 5 chunks in flight).
// ---------------------------------------------------------------------------
constexpr int BM = 64, BN = 256, BK = 32;

struct alignas(16) GemmSmem {     // assumed to sit at LDS offset 0 (sole LDS var)
  bf16 A[2][BM * BK];             // 2 x 4 KB  at offsets 0, 4096
  bf16 Bt[2][BN * BK];            // 2 x 16 KB at offsets 8192, 24576
};
constexpr unsigned LDS_A_OFF = 0;
constexpr unsigned LDS_B_OFF = 2u * BM * BK * 2u;   // 8192

__global__ __launch_bounds__(256) void gemm_bf16_nt(const bf16* __restrict__ A,
                                                    const bf16* __restrict__ Bw,
                                                    float* __restrict__ C,
                                                    int M, int N, int K) {
  __shared__ GemmSmem sm;
  const int tid  = threadIdx.x;
  const int lane = tid & 31;
  const int wave = tid >> 5;
  const int wr   = wave >> 2;    // 0..1
  const int wc   = wave & 3;     // 0..3
  const int m0   = blockIdx.y * BM;
  const int n0   = blockIdx.x * BN;
  const int nk   = K / BK;

  v8f acc[2][4];
#pragma unroll
  for (int s = 0; s < 2; ++s)
#pragma unroll
    for (int t = 0; t < 4; ++t)
#pragma unroll
      for (int i = 0; i < 8; ++i) acc[s][t][i] = 0.0f;

#if HAVE_TDM
  {
    // Visible stores to sm: (a) kills "load of never-stored LDS -> undef",
    // (b) deterministic contents under a mis-programmed descriptor.
    v8bf z = {};
    bf16* base = (bf16*)&sm;
    for (int i = tid * 8; i < (int)(sizeof(GemmSmem) / 2); i += 256 * 8)
      *reinterpret_cast<v8bf*>(base + i) = z;
  }
  __syncthreads();
  // ---- prologue: DMA tiles 0 and 1 (2 descriptors each) from wave 0 ----
  if (wave == 0) {
    tdm_load_2d(LDS_A_OFF,                 &A[(size_t)m0 * K],       K, M, BK, BM, K);
    tdm_load_2d(LDS_B_OFF,                 &Bw[(size_t)n0 * K],      K, N, BK, BN, K);
    tdm_load_2d(LDS_A_OFF + BM * BK * 2u,  &A[(size_t)m0 * K + BK],  K, M, BK, BM, K);
    tdm_load_2d(LDS_B_OFF + BN * BK * 2u,  &Bw[(size_t)n0 * K + BK], K, N, BK, BN, K);
  }
#else
  const int ar = tid >> 2;            // 0..63
  const int ac = (tid & 3) << 3;      // 0,8,16,24 (16B chunks)
  {                                   // prologue: tile 0 via registers
    v8bf ra = *reinterpret_cast<const v8bf*>(&A[(size_t)(m0 + ar) * K + ac]);
    v8bf rb[4];
#pragma unroll
    for (int i = 0; i < 4; ++i) {
      int br = (tid + i * 256) >> 2;
      rb[i] = *reinterpret_cast<const v8bf*>(&Bw[(size_t)(n0 + br) * K + ac]);
    }
    *reinterpret_cast<v8bf*>(&sm.A[0][ar * BK + ac]) = ra;
#pragma unroll
    for (int i = 0; i < 4; ++i) {
      int br = (tid + i * 256) >> 2;
      *reinterpret_cast<v8bf*>(&sm.Bt[0][br * BK + ac]) = rb[i];
    }
  }
  v8bf nra;
  v8bf nrb[4];
#endif

  for (int kt = 0; kt < nk; ++kt) {
    const int buf = kt & 1;

#if HAVE_TDM
    if (wave == 0) {                       // tile kt done; kt+1 may be in flight
      if (kt + 1 < nk) __builtin_amdgcn_s_wait_tensorcnt(2);
      else             __builtin_amdgcn_s_wait_tensorcnt(0);
    }
    __syncthreads();
    lds_publish(&sm);                      // asm may have (re)written sm
#else
    const bool more = (kt + 1 < nk);
    if (more) {                            // issue ALL next-tile loads first
      const int k0 = (kt + 1) * BK;
      nra = *reinterpret_cast<const v8bf*>(&A[(size_t)(m0 + ar) * K + k0 + ac]);
#pragma unroll
      for (int i = 0; i < 4; ++i) {
        int br = (tid + i * 256) >> 2;
        nrb[i] = *reinterpret_cast<const v8bf*>(&Bw[(size_t)(n0 + br) * K + k0 + ac]);
      }
      if (kt + 2 < nk)
        __builtin_prefetch(&A[(size_t)(m0 + ar) * K + (kt + 2) * BK + ac], 0, 1);
    }
    __syncthreads();
#endif

    // ---- fragments + 8 WMMA from LDS buf ----
    v16bf af[2];
#pragma unroll
    for (int s = 0; s < 2; ++s) {
      int row  = wr * 32 + s * 16 + (lane & 15);
      int koff = (lane >> 4) << 3;
      v8bf lo = *reinterpret_cast<const v8bf*>(&sm.A[buf][row * BK + koff]);
      v8bf hi = *reinterpret_cast<const v8bf*>(&sm.A[buf][row * BK + koff + 16]);
      af[s] = make_v16(lo, hi);
    }
    v16bf bfr[4];
#pragma unroll
    for (int t = 0; t < 4; ++t) {
      int col  = wc * 64 + t * 16 + (lane & 15);
      int koff = (lane >> 4) << 3;
      v8bf lo = *reinterpret_cast<const v8bf*>(&sm.Bt[buf][col * BK + koff]);
      v8bf hi = *reinterpret_cast<const v8bf*>(&sm.Bt[buf][col * BK + koff + 16]);
      bfr[t] = make_v16(lo, hi);
    }
#pragma unroll
    for (int s = 0; s < 2; ++s)
#pragma unroll
      for (int t = 0; t < 4; ++t)
        acc[s][t] = WMMA_BF16(af[s], bfr[t], acc[s][t]);

#if HAVE_TDM
    __syncthreads();                       // all waves done reading buf
    if (wave == 0 && kt + 2 < nk) {        // DMA tile kt+2 into buf
      const int k0 = (kt + 2) * BK;
      tdm_load_2d(LDS_A_OFF + (unsigned)buf * BM * BK * 2u,
                  &A[(size_t)m0 * K + k0], K, M, BK, BM, K);
      tdm_load_2d(LDS_B_OFF + (unsigned)buf * BN * BK * 2u,
                  &Bw[(size_t)n0 * K + k0], K, N, BK, BN, K);
    }
#else
    if (more) {                            // park next tile in buf^1
#pragma unroll
      for (int i = 0; i < 4; ++i) {
        int br = (tid + i * 256) >> 2;
        *reinterpret_cast<v8bf*>(&sm.Bt[buf ^ 1][br * BK + ac]) = nrb[i];
      }
      *reinterpret_cast<v8bf*>(&sm.A[buf ^ 1][ar * BK + ac]) = nra;
    }
#endif
  }

  // ---- epilogue: C-layout -> row-major f32 ----
#pragma unroll
  for (int s = 0; s < 2; ++s)
#pragma unroll
    for (int t = 0; t < 4; ++t)
#pragma unroll
      for (int i = 0; i < 8; ++i) {
        int m = m0 + wr * 32 + s * 16 + i + ((lane >> 4) << 3);
        int n = n0 + wc * 64 + t * 16 + (lane & 15);
        C[(size_t)m * N + n] = acc[s][t][i];
      }
}

// ---------------------------------------------------------------------------
// RoPE + unpack: fused f32 [B*S, 6144] -> Q bf16 [B,NH,S,128],
// K bf16 [B,8,S,128] (RoPE'd), V^T bf16 [B,8,128,S].
// ---------------------------------------------------------------------------
__global__ __launch_bounds__(128) void rope_pack(const float* __restrict__ fused,
                                                 const int* __restrict__ pos_ids,
                                                 bf16* __restrict__ Qb,
                                                 bf16* __restrict__ Kb,
                                                 bf16* __restrict__ Vt) {
  const int token = blockIdx.x;
  const int b = token / S_;
  const int s = token - b * S_;
  const int d = threadIdx.x;
  const float* row = fused + (size_t)token * QKV_N;
  const float pos = (float)pos_ids[token];
  const int j = d & 63;
  const float inv_freq = __powf(10000.0f, -(float)j * (1.0f / 64.0f));
  float sn, cs;
  __sincosf(pos * inv_freq, &sn, &cs);

#pragma unroll
  for (int hq = 0; hq < NH; ++hq) {
    const int off = (((hq >> 2) * 6) + (hq & 3)) * HD;
    float x  = row[off + d];
    float xo = (d < 64) ? -row[off + d + 64] : row[off + d - 64];
    Qb[(((size_t)(b * NH + hq) * S_) + s) * HD + d] = (bf16)(x * cs + xo * sn);
  }
#pragma unroll
  for (int gk = 0; gk < NKVH; ++gk) {
    const int koff = (gk * 6 + 4) * HD;
    float x  = row[koff + d];
    float xo = (d < 64) ? -row[koff + d + 64] : row[koff + d - 64];
    Kb[(((size_t)(b * NKVH + gk) * S_) + s) * HD + d] = (bf16)(x * cs + xo * sn);
    const int voff = (gk * 6 + 5) * HD;
    Vt[(((size_t)(b * NKVH + gk) * HD) + d) * S_ + s] = (bf16)row[voff + d];
  }
}

// ---------------------------------------------------------------------------
// Flash attention: block = (b, h, 128-query tile); 8 waves x 16 query rows.
// Causal online softmax; QK^T and PV via WMMA; P re-striped through LDS.
// ---------------------------------------------------------------------------
__global__ __launch_bounds__(256) void flash_attn(const bf16* __restrict__ Qb,
                                                  const bf16* __restrict__ Kb,
                                                  const bf16* __restrict__ Vt,
                                                  bf16* __restrict__ Ctx) {
  const int qblk = blockIdx.x;
  const int h    = blockIdx.y;
  const int b    = blockIdx.z;
  const int g    = h >> 2;
  const int wave = threadIdx.x >> 5;
  const int lane = threadIdx.x & 31;
  const int q0    = qblk * 128;
  const int qrow0 = q0 + wave * 16;

  const bf16* Q  = Qb + ((size_t)(b * NH   + h) * S_) * HD;
  const bf16* Kp = Kb + ((size_t)(b * NKVH + g) * S_) * HD;
  const bf16* Vp = Vt + ((size_t)(b * NKVH + g) * HD) * S_;

  __shared__ bf16 sP[8][16 * 64];

  v16bf qf[4];
#pragma unroll
  for (int kc = 0; kc < 4; ++kc)
    qf[kc] = load_a_frag_g(Q + (size_t)qrow0 * HD, HD, kc * 32);

  v8f acc[8];
#pragma unroll
  for (int dt = 0; dt < 8; ++dt)
#pragma unroll
    for (int i = 0; i < 8; ++i) acc[dt][i] = 0.0f;

  float mrun[8], lrun[8];
#pragma unroll
  for (int i = 0; i < 8; ++i) { mrun[i] = -1e30f; lrun[i] = 0.0f; }

  const float scale = 0.08838834764831845f;   // 1/sqrt(128)
  const int mrow = qrow0 + ((lane >> 4) << 3);

  for (int kb = 0; kb < q0 + 128; kb += 64) {
    v8f st[4];
#pragma unroll
    for (int t = 0; t < 4; ++t) {
      v8f s;
#pragma unroll
      for (int i = 0; i < 8; ++i) s[i] = 0.0f;
#pragma unroll
      for (int kc = 0; kc < 4; ++kc) {
        v16bf kf = load_b_frag_g(Kp + (size_t)(kb + t * 16) * HD, HD, kc * 32);
        s = WMMA_BF16(qf[kc], kf, s);
      }
      st[t] = s;
    }
#pragma unroll
    for (int t = 0; t < 4; ++t) {
      const int n = kb + t * 16 + (lane & 15);
#pragma unroll
      for (int i = 0; i < 8; ++i) {
        float v = st[t][i] * scale;
        st[t][i] = (n > mrow + i) ? -1e30f : v;
      }
    }
    float rmax[8];
#pragma unroll
    for (int i = 0; i < 8; ++i) {
      float v = fmaxf(fmaxf(st[0][i], st[1][i]), fmaxf(st[2][i], st[3][i]));
#pragma unroll
      for (int dlt = 1; dlt < 16; dlt <<= 1) v = fmaxf(v, __shfl_xor(v, dlt, 32));
      rmax[i] = v;
    }
    float alpha[8];
#pragma unroll
    for (int i = 0; i < 8; ++i) {
      float mn = fmaxf(mrun[i], rmax[i]);
      alpha[i] = __expf(mrun[i] - mn);
      mrun[i] = mn;
    }
    float rsum[8];
#pragma unroll
    for (int i = 0; i < 8; ++i) rsum[i] = 0.0f;
#pragma unroll
    for (int t = 0; t < 4; ++t)
#pragma unroll
      for (int i = 0; i < 8; ++i) {
        float e = __expf(st[t][i] - mrun[i]);
        st[t][i] = e;
        rsum[i] += e;
      }
#pragma unroll
    for (int i = 0; i < 8; ++i) {
      float v = rsum[i];
#pragma unroll
      for (int dlt = 1; dlt < 16; dlt <<= 1) v += __shfl_xor(v, dlt, 32);
      lrun[i] = lrun[i] * alpha[i] + v;
    }
#pragma unroll
    for (int dt = 0; dt < 8; ++dt)
#pragma unroll
      for (int i = 0; i < 8; ++i) acc[dt][i] *= alpha[i];

#pragma unroll
    for (int t = 0; t < 4; ++t) {
      const int n = t * 16 + (lane & 15);
#pragma unroll
      for (int i = 0; i < 8; ++i) {
        const int r = i + ((lane >> 4) << 3);
        sP[wave][r * 64 + n] = (bf16)st[t][i];
      }
    }
    v16bf pf[2];
#pragma unroll
    for (int pc = 0; pc < 2; ++pc) {
      const int row  = lane & 15;
      const int koff = (lane >> 4) << 3;
      const bf16* p = &sP[wave][row * 64 + pc * 32 + koff];
      v8bf lo = *reinterpret_cast<const v8bf*>(p);
      v8bf hi = *reinterpret_cast<const v8bf*>(p + 16);
      pf[pc] = make_v16(lo, hi);
    }
#pragma unroll
    for (int dt = 0; dt < 8; ++dt)
#pragma unroll
      for (int pc = 0; pc < 2; ++pc) {
        v16bf vf = load_b_frag_g(Vp + (size_t)(dt * 16) * S_, S_, kb + pc * 32);
        acc[dt] = WMMA_BF16(pf[pc], vf, acc[dt]);
      }
  }

#pragma unroll
  for (int dt = 0; dt < 8; ++dt)
#pragma unroll
    for (int i = 0; i < 8; ++i) {
      const int m = qrow0 + i + ((lane >> 4) << 3);
      const int d = dt * 16 + (lane & 15);
      const float o = acc[dt][i] / lrun[i];
      Ctx[((size_t)(b * S_ + m)) * (NH * HD) + h * HD + d] = (bf16)o;
    }
}

// ---------------------------------------------------------------------------
// Launch
// ---------------------------------------------------------------------------
extern "C" void kernel_launch(void* const* d_in, const int* in_sizes, int n_in,
                              void* d_out, int out_size, void* d_ws, size_t ws_size,
                              hipStream_t stream) {
  (void)in_sizes; (void)n_in; (void)out_size; (void)ws_size;
  const float* hidden  = (const float*)d_in[0];
  const int*   posids  = (const int*)d_in[1];
  const float* w_qkv   = (const float*)d_in[2];
  const float* w_dense = (const float*)d_in[3];
  float* out = (float*)d_out;

  // Workspace layout (bytes); total = 301,989,888 (~288 MiB)
  char* ws = (char*)d_ws;
  bf16*  Xbf   = (bf16*)(ws + 0);              // 4096x4096 bf16   (32 MiB)
  bf16*  Wqkv  = (bf16*)(ws + 33554432);       // 6144x4096 bf16   (48 MiB)
  bf16*  Wd    = (bf16*)(ws + 83886080);       // 4096x4096 bf16   (32 MiB)
  float* fused = (float*)(ws + 117440512);     // 4096x6144 f32    (96 MiB)
  bf16*  Qb    = (bf16*)(ws + 218103808);      // 2x32x2048x128    (32 MiB)
  bf16*  Kb    = (bf16*)(ws + 251658240);      // 2x8x2048x128     (8 MiB)
  bf16*  Vt    = (bf16*)(ws + 260046848);      // 2x8x128x2048     (8 MiB)
  bf16*  Ctx   = (bf16*)(ws + 268435456);      // 4096x4096 bf16   (32 MiB)

  cvt_f32_bf16<<<(M_TOK * H_) / 1024, 256, 0, stream>>>(hidden, Xbf, M_TOK * H_);
  cvt_f32_bf16<<<(QKV_N * H_) / 1024, 256, 0, stream>>>(w_qkv, Wqkv, QKV_N * H_);
  cvt_f32_bf16<<<(H_ * H_) / 1024, 256, 0, stream>>>(w_dense, Wd, H_ * H_);

  dim3 g1(QKV_N / BN, M_TOK / BM);             // (24, 64)
  gemm_bf16_nt<<<g1, 256, 0, stream>>>(Xbf, Wqkv, fused, M_TOK, QKV_N, H_);

  rope_pack<<<M_TOK, 128, 0, stream>>>(fused, posids, Qb, Kb, Vt);

  dim3 g2(S_ / 128, NH, B_);                   // (16, 32, 2)
  flash_attn<<<g2, 256, 0, stream>>>(Qb, Kb, Vt, Ctx);

  dim3 g3(H_ / BN, M_TOK / BM);                // (16, 64)
  gemm_bf16_nt<<<g3, 256, 0, stream>>>(Ctx, Wd, out, M_TOK, H_, H_);
}
